// DGCNN_gpvn_purenet_74071005987492
// MI455X (gfx1250) — matrix operand order repaired and checked by reading, not compiled, not run on hardware
//
#include <hip/hip_runtime.h>

// ---------------------------------------------------------------------------
// DGCNN forward for MI455X (gfx1250).  All 1x1 convs run as bf16 WMMA GEMMs
// (v_wmma_f32_16x16x32_bf16, fp32 accumulate); concat-broadcast inputs of the
// last two convs are fused into the GEMM tile loader to avoid materializing a
// 335MB tensor.  wave32/WGP-aware: 256-thread blocks = 8 waves; block tile
// 64(o) x 64(p); each wave owns 32x16 output (two 16x16 WMMA accumulators
// sharing one B fragment).  X tile is staged transposed (p-major) so both
// fragment types assemble with aligned ds_load_b32.  The k-loop is software
// pipelined: tile t+1 is fetched to registers before tile t's WMMAs issue.
// ---------------------------------------------------------------------------

typedef __attribute__((ext_vector_type(16))) __bf16 v16bf;
typedef __attribute__((ext_vector_type(8)))  float  v8f;

#define BATCH 2
#define NPTS  1024
#define KNN_K 20
#define KPTS  64
#define SLOPE 0.2f
#define GN_EPS 1e-5f

static __device__ __forceinline__ float lrelu(float v) { return v >= 0.f ? v : SLOPE * v; }

// ---------------------------------------------------------------------------
// Brute-force KNN (top-20 of negative squared distance, self included).
// ---------------------------------------------------------------------------
__global__ void knn_kernel(const float* __restrict__ X, int* __restrict__ IDX,
                           int C, int bstride) {
  int gid = blockIdx.x * blockDim.x + threadIdx.x;
  if (gid >= BATCH * NPTS) return;
  int b = gid / NPTS, n = gid % NPTS;
  const float* xb = X + (size_t)b * bstride;
  float xn[64];
  float xx = 0.f;
  for (int c = 0; c < C; ++c) { float v = xb[c * NPTS + n]; xn[c] = v; xx += v * v; }
  float val[KNN_K]; int id[KNN_K];
#pragma unroll
  for (int j = 0; j < KNN_K; ++j) { val[j] = -3.4e38f; id[j] = 0; }
  for (int m = 0; m < NPTS; ++m) {
    float dot = 0.f, xm = 0.f;
    for (int c = 0; c < C; ++c) { float v = xb[c * NPTS + m]; dot += xn[c] * v; xm += v * v; }
    float pd = 2.f * dot - xx - xm;
    if (pd > val[KNN_K - 1]) {
      int pos = KNN_K - 1;
#pragma unroll
      for (int j = KNN_K - 2; j >= 0; --j)
        if (pd > val[j]) { val[j + 1] = val[j]; id[j + 1] = id[j]; pos = j; }
      val[pos] = pd; id[pos] = m;
    }
  }
#pragma unroll
  for (int j = 0; j < KNN_K; ++j) IDX[(size_t)gid * KNN_K + j] = id[j];
}

// ---------------------------------------------------------------------------
// graph_feature: F[b, c2, n, k] = (c2<C) ? x[c2, idx]-x[c2, n] : x[c2-C, n]
// ---------------------------------------------------------------------------
__global__ void graph_feature_kernel(const float* __restrict__ X, const int* __restrict__ IDX,
                                     float* __restrict__ F, int C, int bstride) {
  size_t total = (size_t)BATCH * 2 * C * NPTS * KNN_K;
  size_t i = (size_t)blockIdx.x * blockDim.x + threadIdx.x;
  if (i >= total) return;
  int k = (int)(i % KNN_K);
  size_t r = i / KNN_K;
  int n = (int)(r % NPTS); r /= NPTS;
  int c2 = (int)(r % (2 * C));
  int b  = (int)(r / (2 * C));
  const float* xb = X + (size_t)b * bstride;
  int j = IDX[((size_t)b * NPTS + n) * KNN_K + k];
  float v = (c2 < C) ? (xb[c2 * NPTS + j] - xb[c2 * NPTS + n])
                     : xb[(c2 - C) * NPTS + n];
  F[i] = v;
}

// ---------------------------------------------------------------------------
// WMMA GEMM: Y[b,o,p] = sum_c W[o,c] * X[b,c,p].
// MODE 0: X0 is (B, C, P) contiguous.
// MODE 1: fused concat-broadcast: c <  C0 -> X0[b, c, p>>6]    ((B,C0,N))
//                                 c >= C0 -> X1[b, c-C0, p&63] ((B,C-C0,64))
// Tiles: Wt 64(o) x 32(c) o-major;  Xt 64(p) x 32(c) p-major (transposed).
// Ragged-C handling branches on whole-tile fullness (wave-uniform: scalar
// branch, no exec masking); clamp+select code runs only for the last tile
// of the single C=12 layer.
// ---------------------------------------------------------------------------
template <int MODE>
__global__ void __launch_bounds__(256)
gemm_wmma(const float* __restrict__ W, const float* __restrict__ X0,
          const float* __restrict__ X1, float* __restrict__ Y,
          int O, int C, int P, int C0) {
  __shared__ __align__(16) __bf16 Wt[64 * 34];   // row stride 34 (even -> b32-aligned pairs)
  __shared__ __align__(16) __bf16 Xt[64 * 34];
  const int t    = threadIdx.x;
  const int b    = blockIdx.z;
  const int o0   = blockIdx.y * 64;
  const int p0   = blockIdx.x * 64;
  const int lane = t & 31;
  const int wave = t >> 5;
  const int wr   = (wave >> 2) * 32;   // wave o offset: 0 or 32
  const int wc   = (wave & 3) * 16;    // wave p offset: 0,16,32,48
  const int m    = lane & 15;
  const int h    = lane >> 4;
  v8f acc0 = {}, acc1 = {};

  const int wol = t >> 2;              // W staging row   (0..63)
  const int wcl = (t & 3) * 8;         // W staging col*8
  const int xcl = t >> 3;              // X staging c     (0..31)
  const int xpl = (t & 7) * 8;         // X staging p*8

  union Pack8  { __bf16 e[8]; uint4 q; };
  union FragBF { v16bf v; unsigned int u[8]; };

  const int NT = (C + 31) >> 5;        // number of 32-wide k tiles
  const float* wrow = W + (size_t)(o0 + wol) * C;
  float wf[8], xf[8];                  // register staging (pipeline)

  // fetch one k-tile (global -> registers); uniform branch on fullness
  auto fetchTile = [&](int c0) {
    if (c0 + 32 <= C) {                               // full tile: vector loads
      float4 f0 = *(const float4*)(wrow + c0 + wcl);
      float4 f1 = *(const float4*)(wrow + c0 + wcl + 4);
      wf[0] = f0.x; wf[1] = f0.y; wf[2] = f0.z; wf[3] = f0.w;
      wf[4] = f1.x; wf[5] = f1.y; wf[6] = f1.z; wf[7] = f1.w;
    } else {                                          // ragged tail (C=12 layer only)
#pragma unroll
      for (int j = 0; j < 8; ++j) {
        int c  = c0 + wcl + j;
        int cc = c < C ? c : C - 1;
        float wv = wrow[cc];                          // always-legal load, then select
        wf[j] = (c < C) ? wv : 0.f;
      }
    }
    if (MODE == 0) {
      if (c0 + 32 <= C) {
        const float* xrow = X0 + ((size_t)b * C + c0 + xcl) * P + p0 + xpl;
#pragma unroll
        for (int j = 0; j < 8; ++j) xf[j] = xrow[j];  // contiguous -> b128s
      } else {
        int c  = c0 + xcl;
        int cc = (c < C) ? c : 0;
        const float* xrow = X0 + ((size_t)b * C + cc) * P + p0 + xpl;
#pragma unroll
        for (int j = 0; j < 8; ++j) { float v = xrow[j]; xf[j] = (c < C) ? v : 0.f; }
      }
    } else {                                          // MODE1: C % 32 == 0 always
      int c = c0 + xcl;
      if (c < C0) {                                   // h7 path: p>>6 constant in chunk
        float v = X0[((size_t)b * C0 + c) * NPTS + ((p0 + xpl) >> 6)];
#pragma unroll
        for (int j = 0; j < 8; ++j) xf[j] = v;
      } else {
        const float* xrow = X1 + ((size_t)b * (C - C0) + (c - C0)) * KPTS + (xpl & (KPTS - 1));
#pragma unroll
        for (int j = 0; j < 8; ++j) xf[j] = xrow[j];
      }
    }
  };

  // registers -> LDS (W packed as one b128 store; X transposed b16 stores)
  auto storeTile = [&]() {
    Pack8 pk;
#pragma unroll
    for (int j = 0; j < 8; ++j) pk.e[j] = (__bf16)wf[j];
    *(uint4*)(Wt + wol * 34 + wcl) = pk.q;
#pragma unroll
    for (int j = 0; j < 8; ++j) Xt[(xpl + j) * 34 + xcl] = (__bf16)xf[j];
  };

  fetchTile(0);
  for (int it = 0; it < NT; ++it) {
    storeTile();
    __syncthreads();
    if (it + 1 < NT) fetchTile((it + 1) * 32);        // pipeline: loads fly over WMMAs
    if (it + 2 < NT)                                  // global_prefetch_b8 two tiles ahead
      __builtin_prefetch(wrow + (it + 2) * 32 + wcl, 0, 1);

    FragBF av0, av1, bv;
#pragma unroll
    for (int v = 0; v < 8; ++v) {
      int ka = ((v & 3) * 2) + ((v & 4) ? 16 : 0) + h * 8;   // A K-pair
      av0.u[v] = *(const unsigned int*)(Wt + (wr + m) * 34 + ka);
      av1.u[v] = *(const unsigned int*)(Wt + (wr + 16 + m) * 34 + ka);
      int kb = ((v & 3) * 2) + ((v & 4) ? 16 : 0) + h * 8;   // B K-pair
      bv.u[v]  = *(const unsigned int*)(Xt + (wc + m) * 34 + kb);
    }
    acc0 = __builtin_amdgcn_wmma_f32_16x16x32_bf16(false, av0.v, false, bv.v,
                                                   (short)0, acc0, false, false);
    acc1 = __builtin_amdgcn_wmma_f32_16x16x32_bf16(false, av1.v, false, bv.v,
                                                   (short)0, acc1, false, false);
    __syncthreads();
  }

  // C/D layout: VGPR r -> M = r + 8*h ; N = lane&15
  const int p = p0 + wc + m;
#pragma unroll
  for (int r = 0; r < 8; ++r) {
    int oA = o0 + wr + r + h * 8;
    Y[((size_t)b * O + oA) * P + p]      = acc0[r];
    Y[((size_t)b * O + oA + 16) * P + p] = acc1[r];
  }
}

// ---------------------------------------------------------------------------
// GroupNorm (32 groups): stats per (b, g) over contiguous (C/32)*S block.
// ---------------------------------------------------------------------------
__global__ void gn_stats_kernel(const float* __restrict__ X, float* __restrict__ stats,
                                int C, int S) {
  int bg = blockIdx.x;                         // b*32 + g
  int cg = C >> 5;
  size_t total = (size_t)cg * S;
  const float* base = X + (size_t)bg * total;  // group data is contiguous
  float s = 0.f, ss = 0.f;
  for (size_t i = threadIdx.x; i < total; i += blockDim.x) {
    float v = base[i]; s += v; ss += v * v;
  }
  __shared__ float sh0[256], sh1[256];
  sh0[threadIdx.x] = s; sh1[threadIdx.x] = ss;
  __syncthreads();
  for (int st = 128; st > 0; st >>= 1) {
    if ((int)threadIdx.x < st) {
      sh0[threadIdx.x] += sh0[threadIdx.x + st];
      sh1[threadIdx.x] += sh1[threadIdx.x + st];
    }
    __syncthreads();
  }
  if (threadIdx.x == 0) {
    float inv = 1.f / (float)total;
    float mu  = sh0[0] * inv;
    float var = sh1[0] * inv - mu * mu;
    stats[bg * 2]     = mu;
    stats[bg * 2 + 1] = rsqrtf(var + GN_EPS);
  }
}

__global__ void gn_apply_kernel(float* __restrict__ X, const float* __restrict__ stats,
                                const float* __restrict__ gamma, const float* __restrict__ beta,
                                int C, int S) {
  size_t total = (size_t)BATCH * C * S;
  size_t i = (size_t)blockIdx.x * blockDim.x + threadIdx.x;
  if (i >= total) return;
  int cg = C >> 5;
  size_t cs = (size_t)C * S;
  int b = (int)(i / cs);
  size_t r = i % cs;
  int c = (int)(r / S);
  int g = c / cg;
  float mu = stats[(b * 32 + g) * 2];
  float rs = stats[(b * 32 + g) * 2 + 1];
  float v = (X[i] - mu) * rs * gamma[c] + beta[c];
  X[i] = lrelu(v);
}

// max over last dim L of (B,C,NPTS,L) -> dst[b*outB + (cofs+c)*NPTS + n]
__global__ void max_last_kernel(const float* __restrict__ X, float* __restrict__ Y,
                                int C, int L, int outBstride, int cofs) {
  int total = BATCH * C * NPTS;
  int i = blockIdx.x * blockDim.x + threadIdx.x;
  if (i >= total) return;
  int n = i % NPTS;
  int c = (i / NPTS) % C;
  int b = i / (C * NPTS);
  const float* p = X + (((size_t)b * C + c) * NPTS + n) * L;
  float mx = p[0];
  for (int k = 1; k < L; ++k) mx = fmaxf(mx, p[k]);
  Y[(size_t)b * outBstride + (size_t)(cofs + c) * NPTS + n] = mx;
}

// global max over n of H6 (B,1024,NPTS) -> G (B,1024)
__global__ void max_n_kernel(const float* __restrict__ X, float* __restrict__ Y) {
  int i = blockIdx.x * blockDim.x + threadIdx.x;
  if (i >= BATCH * 1024) return;
  const float* p = X + (size_t)i * NPTS;
  float mx = p[0];
  for (int n = 1; n < NPTS; ++n) mx = fmaxf(mx, p[n]);
  Y[i] = mx;
}

// CAT (B,1216,NPTS) = [g bcast | x1 | x2 | x3]
__global__ void cat1216_kernel(const float* __restrict__ G, const float* __restrict__ X123,
                               float* __restrict__ CAT) {
  size_t total = (size_t)BATCH * 1216 * NPTS;
  size_t i = (size_t)blockIdx.x * blockDim.x + threadIdx.x;
  if (i >= total) return;
  int n = (int)(i % NPTS);
  int c = (int)((i / NPTS) % 1216);
  int b = (int)(i / ((size_t)1216 * NPTS));
  CAT[i] = (c < 1024) ? G[b * 1024 + c]
                      : X123[((size_t)b * 192 + (c - 1024)) * NPTS + n];
}

// x_seg[b, 0/1, n] = w9 @ max_kp(h8r)   (max precomputed into M8r)
__global__ void seg_kernel(const float* __restrict__ M8r, const float* __restrict__ W9,
                           float* __restrict__ OUT) {
  int i = blockIdx.x * blockDim.x + threadIdx.x;
  if (i >= BATCH * NPTS) return;
  int b = i / NPTS, n = i % NPTS;
  float a0 = 0.f, a1 = 0.f;
  for (int o = 0; o < 256; ++o) {
    float v = M8r[((size_t)b * 256 + o) * NPTS + n];
    a0 += W9[o] * v;
    a1 += W9[256 + o] * v;
  }
  OUT[(size_t)b * 2 * NPTS + n]        = a0;
  OUT[(size_t)b * 2 * NPTS + NPTS + n] = a1;
}

// x_key[b, kp, n] = sum_o w10[o] * h8r[b, o, n, kp]
__global__ void key_kernel(const float* __restrict__ H8, const float* __restrict__ W10,
                           float* __restrict__ OUT) {
  int i = blockIdx.x * blockDim.x + threadIdx.x;
  if (i >= BATCH * KPTS * NPTS) return;
  int n  = i % NPTS;
  int kp = (i / NPTS) % KPTS;
  int b  = i / (KPTS * NPTS);
  float a = 0.f;
  for (int o = 0; o < 256; ++o)
    a += W10[o] * H8[(((size_t)b * 256 + o) * NPTS + n) * KPTS + kp];
  OUT[i] = a;
}

// ---------------------------------------------------------------------------
extern "C" void kernel_launch(void* const* d_in, const int* in_sizes, int n_in,
                              void* d_out, int out_size, void* d_ws, size_t ws_size,
                              hipStream_t stream) {
  (void)in_sizes; (void)n_in; (void)out_size; (void)ws_size;
  const float* x   = (const float*)d_in[0];
  const float* obj = (const float*)d_in[1];
  const float* w1  = (const float*)d_in[2];  const float* g1  = (const float*)d_in[3];  const float* b1  = (const float*)d_in[4];
  const float* w2  = (const float*)d_in[5];  const float* g2  = (const float*)d_in[6];  const float* b2  = (const float*)d_in[7];
  const float* w3  = (const float*)d_in[8];  const float* g3  = (const float*)d_in[9];  const float* b3  = (const float*)d_in[10];
  const float* w4  = (const float*)d_in[11]; const float* g4  = (const float*)d_in[12]; const float* b4  = (const float*)d_in[13];
  const float* w5  = (const float*)d_in[14]; const float* g5  = (const float*)d_in[15]; const float* b5  = (const float*)d_in[16];
  const float* w6  = (const float*)d_in[17]; const float* g6  = (const float*)d_in[18]; const float* b6  = (const float*)d_in[19];
  const float* w7  = (const float*)d_in[20]; const float* g7  = (const float*)d_in[21]; const float* b7  = (const float*)d_in[22];
  const float* w8  = (const float*)d_in[23]; const float* g8  = (const float*)d_in[24]; const float* b8  = (const float*)d_in[25];
  const float* w8r = (const float*)d_in[26]; const float* g8r = (const float*)d_in[27]; const float* b8r = (const float*)d_in[28];
  const float* w9  = (const float*)d_in[29];
  const float* w10 = (const float*)d_in[30];
  float* out = (float*)d_out;

  // ---- workspace carve (floats), ~193 MB peak ----
  float* ws = (float*)d_ws;
  size_t off = 0;
  int*   IDX  = (int*)(ws + off); off += (size_t)BATCH * NPTS * KNN_K;
  float* A    = ws + off;         off += (size_t)BATCH * 128 * NPTS * KNN_K;  // edge feat / ping
  float* Bb   = ws + off;         off += (size_t)BATCH * 64  * NPTS * KNN_K;  // pong
  float* X123 = ws + off;         off += (size_t)BATCH * 192 * NPTS;          // [x1|x2|x3]
  float* H6   = ws + off;         off += (size_t)BATCH * 1024 * NPTS;
  float* G    = ws + off;         off += (size_t)BATCH * 1024;
  float* CAT  = ws + off;         off += (size_t)BATCH * 1216 * NPTS;
  float* H7   = ws + off;         off += (size_t)BATCH * 512 * NPTS;
  float* H8   = ws + off;         off += (size_t)BATCH * 256 * NPTS * KPTS;   // reused: h8 then h8r
  float* M8   = ws + off;         off += (size_t)BATCH * 256 * NPTS;          // reused max buffers
  float* ST   = ws + off;         off += (size_t)BATCH * 32 * 2;              // GN stats

  const int TB = 256;
  const int NK = NPTS * KNN_K;   // 20480
  auto gridFor = [](size_t total, int bs) { return dim3((unsigned)((total + bs - 1) / bs)); };

  auto conv_gn = [&](const float* Wp, const float* X0p, const float* X1p, float* Yp,
                     int O, int C, int P, int C0, const float* gm, const float* bt, int mode) {
    dim3 g(P / 64, O / 64, BATCH);
    if (mode == 0)
      gemm_wmma<0><<<g, 256, 0, stream>>>(Wp, X0p, nullptr, Yp, O, C, P, 0);
    else
      gemm_wmma<1><<<g, 256, 0, stream>>>(Wp, X0p, X1p, Yp, O, C, P, C0);
    gn_stats_kernel<<<BATCH * 32, 256, 0, stream>>>(Yp, ST, O, P);
    gn_apply_kernel<<<gridFor((size_t)BATCH * O * P, TB), TB, 0, stream>>>(Yp, ST, gm, bt, O, P);
  };

  // ---- EdgeConv block 1 (KNN on xyz only) ----
  knn_kernel<<<gridFor(BATCH * NPTS, TB), TB, 0, stream>>>(x, IDX, 3, 6 * NPTS);
  graph_feature_kernel<<<gridFor((size_t)BATCH * 12 * NK, TB), TB, 0, stream>>>(x, IDX, A, 6, 6 * NPTS);
  conv_gn(w1, A, nullptr, Bb, 64, 12, NK, 0, g1, b1, 0);
  conv_gn(w2, Bb, nullptr, A, 64, 64, NK, 0, g2, b2, 0);
  max_last_kernel<<<gridFor(BATCH * 64 * NPTS, TB), TB, 0, stream>>>(A, X123, 64, KNN_K, 192 * NPTS, 0);

  // ---- EdgeConv block 2 ----
  knn_kernel<<<gridFor(BATCH * NPTS, TB), TB, 0, stream>>>(X123, IDX, 64, 192 * NPTS);
  graph_feature_kernel<<<gridFor((size_t)BATCH * 128 * NK, TB), TB, 0, stream>>>(X123, IDX, A, 64, 192 * NPTS);
  conv_gn(w3, A, nullptr, Bb, 64, 128, NK, 0, g3, b3, 0);
  conv_gn(w4, Bb, nullptr, A, 64, 64, NK, 0, g4, b4, 0);
  max_last_kernel<<<gridFor(BATCH * 64 * NPTS, TB), TB, 0, stream>>>(A, X123, 64, KNN_K, 192 * NPTS, 64);

  // ---- EdgeConv block 3 ----
  knn_kernel<<<gridFor(BATCH * NPTS, TB), TB, 0, stream>>>(X123 + 64 * NPTS, IDX, 64, 192 * NPTS);
  graph_feature_kernel<<<gridFor((size_t)BATCH * 128 * NK, TB), TB, 0, stream>>>(X123 + 64 * NPTS, IDX, A, 64, 192 * NPTS);
  conv_gn(w5, A, nullptr, Bb, 64, 128, NK, 0, g5, b5, 0);
  max_last_kernel<<<gridFor(BATCH * 64 * NPTS, TB), TB, 0, stream>>>(Bb, X123, 64, KNN_K, 192 * NPTS, 128);

  // ---- global feature path ----
  conv_gn(w6, X123, nullptr, H6, 1024, 192, NPTS, 0, g6, b6, 0);
  max_n_kernel<<<gridFor(BATCH * 1024, TB), TB, 0, stream>>>(H6, G);
  cat1216_kernel<<<gridFor((size_t)BATCH * 1216 * NPTS, TB), TB, 0, stream>>>(G, X123, CAT);
  conv_gn(w7, CAT, nullptr, H7, 512, 1216, NPTS, 0, g7, b7, 0);

  // ---- keypoint heads (fused concat-broadcast GEMMs) ----
  conv_gn(w8, H7, obj, H8, 256, 640, NPTS * KPTS, 512, g8, b8, 1);
  max_last_kernel<<<gridFor(BATCH * 256 * NPTS, TB), TB, 0, stream>>>(H8, M8, 256, KPTS, 256 * NPTS, 0);
  conv_gn(w8r, M8, obj, H8, 256, 384, NPTS * KPTS, 256, g8r, b8r, 1);
  max_last_kernel<<<gridFor(BATCH * 256 * NPTS, TB), TB, 0, stream>>>(H8, M8, 256, KPTS, 256 * NPTS, 0);

  // ---- outputs: x_seg (B,2,N) then x_key (B,KP,N) ----
  seg_kernel<<<gridFor(BATCH * NPTS, TB), TB, 0, stream>>>(M8, w9, out);
  key_kernel<<<gridFor(BATCH * KPTS * NPTS, TB), TB, 0, stream>>>(H8, w10, out + (size_t)BATCH * 2 * NPTS);
}
